// VCGDH_V_78477642432532
// MI455X (gfx1250) — compile-verified, matrix operand order checked by reading, not run on hardware
//
#include <hip/hip_runtime.h>
#include <hip/hip_bf16.h>
#include <math.h>

// =====================================================================
// VCGDH forward for MI455X (gfx1250, wave32, WMMA 16x16x32 bf16)
// One LDS-double-buffered WMMA GEMM template drives all matrix math:
// 5 convs (implicit im2col GEMM, compile-time geometry), FC, f@f^T,
// 2 GCN layers, hash + classifier heads. bf16 operands, f32 accumulate.
// 128x64 C-tile, 8 wave32 waves in a 4x2 grid, 2x2 WMMA tiles per wave.
// =====================================================================

typedef __attribute__((ext_vector_type(16))) __bf16 v16bf;
typedef __attribute__((ext_vector_type(8)))  __bf16 v8bf;
typedef __attribute__((ext_vector_type(8)))  float  v8f;

#define LDS_LD 40   // 32 K-elements + 8 pad (80B rows, 16B-aligned chunks)

// ---------------- A-matrix loaders (prep(m) once, load8 per K-chunk) --
struct ADense {
  const __bf16* p; int K;
  struct Row { const __bf16* ptr; };
  __device__ Row prep(int m) const { return { p + (size_t)m * K }; }
  __device__ v8bf load8(const Row& r, int k) const {
    if (k + 8 <= K) return *(const v8bf*)(r.ptr + k);   // aligned b128
    v8bf v = {};
#pragma unroll
    for (int j = 0; j < 8; ++j) if (k + j < K) v[j] = r.ptr[k + j];
    return v;
  }
};
// implicit im2col with compile-time geometry: all divisions are by
// constants (strength-reduced); k -> (ci,kh,kw) stepped incrementally.
template <int CI, int H, int W, int HO, int WO, int KH, int STRIDE, int PAD>
struct AConvT {
  const __bf16* in;
  struct Row { int b, ihb, iwb; };
  __device__ Row prep(int m) const {
    constexpr int HWo = HO * WO;
    int b  = m / HWo; int r  = m - b * HWo;
    int oh = r / WO;  int ow = r - oh * WO;
    return { b, oh * STRIDE - PAD, ow * STRIDE - PAD };
  }
  __device__ v8bf load8(const Row& r, int k) const {
    constexpr int KK = KH * KH;
    constexpr int KT = CI * KK;
    v8bf v = {};
    int ci = k / KK;  int rk = k - ci * KK;
    int kh = rk / KH; int kw = rk - kh * KH;
    const __bf16* img = in + (size_t)r.b * (CI * H * W);
#pragma unroll
    for (int j = 0; j < 8; ++j) {
      if (k + j < KT) {
        int ih = r.ihb + kh, iw = r.iwb + kw;
        if (ih >= 0 && ih < H && iw >= 0 && iw < W)
          v[j] = img[(ci * H + ih) * W + iw];
      }
      if (++kw == KH) { kw = 0; if (++kh == KH) { kh = 0; ++ci; } }
    }
    return v;
  }
};

// ---------------- B-matrix loaders (GEMM computes A @ B^T) ------------
struct BNK {   // W[n][k] row-major with padded row stride ld (16-elem mult)
  const __bf16* w; int K, ld;
  struct Row { const __bf16* ptr; };
  __device__ Row prep(int n) const { return { w + (size_t)n * ld }; }
  __device__ v8bf load8(const Row& r, int k) const {
    if (k + 8 <= ld) return *(const v8bf*)(r.ptr + k);  // zero-padded rows
    v8bf v = {};
#pragma unroll
    for (int j = 0; j < 8; ++j) if (k + j < K) v[j] = r.ptr[k + j];
    return v;
  }
};
struct BKN {   // X[k][n] (non-transposed right operand), strided gather
  const __bf16* w; int N, K;
  struct Row { const __bf16* ptr; };
  __device__ Row prep(int n) const { return { w + n }; }
  __device__ v8bf load8(const Row& r, int k) const {
    v8bf v = {};
#pragma unroll
    for (int j = 0; j < 8; ++j) if (k + j < K) v[j] = r.ptr[(size_t)(k + j) * N];
    return v;
  }
};

// ---------------- store epilogues; ACT: 0=none 1=relu 2=sigmoid -------
template <typename OutT, int ACT>
struct SDense {
  OutT* out; const float* bias; int N;
  __device__ void put(int m, int n, float v) const {
    if (bias) v += bias[n];
    if (ACT == 1) v = fmaxf(v, 0.0f);
    if (ACT == 2) v = 1.0f / (1.0f + expf(-v));
    out[(size_t)m * N + n] = (OutT)v;
  }
};
template <int CO, int HOWO>      // bias + relu, scatter to NCHW
struct SConvT {
  __bf16* out; const float* bias;
  __device__ void put(int m, int n, float v) const {
    v = fmaxf(v + bias[n], 0.0f);
    int b = m / HOWO; int r = m - b * HOWO;
    out[((size_t)b * CO + n) * HOWO + r] = (__bf16)v;
  }
};

// ---- 16x32 bf16 fragment from LDS per ISA 16-bit layout --------------
// lanes 0-15 (kc=0): K chunks [0..7],[16..23]; lanes 16-31: [8..15],[24..31]
__device__ inline v16bf frag_from_lds(const __bf16* sm, int row, int kc) {
  const __bf16* base = sm + row * LDS_LD;
  v8bf lo = *(const v8bf*)(base + kc * 8);
  v8bf hi = *(const v8bf*)(base + 16 + kc * 8);
  v16bf f;
#pragma unroll
  for (int i = 0; i < 8; ++i) { f[i] = lo[i]; f[i + 8] = hi[i]; }
  return f;
}

// ---- 128x64-tile WMMA GEMM, 8 wave32 waves (4x2), LDS double-buffered
template <class AL, class BL, class CS>
__global__ __launch_bounds__(256)
void gemm_wmma(AL Am, BL Bm, CS Cs, int M, int N, int K) {
  __shared__ __bf16 As[2][128 * LDS_LD];   // 20.0 KB
  __shared__ __bf16 Bs[2][64 * LDS_LD];    // 10.0 KB

  const int tid  = threadIdx.x;
  const int lane = tid & 31;
  const int wave = tid >> 5;
  const int wm   = wave >> 1;       // 0..3 -> m 32-block
  const int wn   = wave & 1;        // 0..1 -> n 32-block
  const int blkM = blockIdx.y * 128;
  const int blkN = blockIdx.x * 64;

  // staging: each thread stages 16 A elems (1 row, 2 chunks) + 8 B elems
  const int arow = tid >> 1;            // 0..127
  const int acol = (tid & 1) * 16;      // 0 | 16
  const int brow = tid >> 2;            // 0..63
  const int bcol = (tid & 3) * 8;       // 0,8,16,24

  const int m0 = wm * 32;
  const int n0 = wn * 32;
  const int kc = lane >> 4;
  const int lr = lane & 15;

  const bool amv = (blkM + arow) < M;
  const bool bnv = (blkN + brow) < N;
  typename AL::Row ar = Am.prep(amv ? (blkM + arow) : (M > 0 ? M - 1 : 0));
  typename BL::Row br = Bm.prep(bnv ? (blkN + brow) : (N > 0 ? N - 1 : 0));

  auto stage = [&](int ks) {
    int kb = ks * 32;
    int bi = ks & 1;
    v8bf z = {};
    v8bf a0 = amv ? Am.load8(ar, kb + acol)     : z;
    v8bf a1 = amv ? Am.load8(ar, kb + acol + 8) : z;
    v8bf bv = bnv ? Bm.load8(br, kb + bcol)     : z;
    *(v8bf*)&As[bi][arow * LDS_LD + acol]     = a0;
    *(v8bf*)&As[bi][arow * LDS_LD + acol + 8] = a1;
    *(v8bf*)&Bs[bi][brow * LDS_LD + bcol]     = bv;
  };

  v8f acc00 = {}, acc01 = {}, acc10 = {}, acc11 = {};
  const int nk = (K + 31) / 32;

  stage(0);
  for (int ks = 0; ks < nk; ++ks) {
    __syncthreads();
    if (ks + 1 < nk) stage(ks + 1);          // overlap: fetch next chunk
    const __bf16* a_s = As[ks & 1];
    const __bf16* b_s = Bs[ks & 1];
    v16bf b0 = frag_from_lds(b_s, n0 + lr, kc);
    v16bf b1 = frag_from_lds(b_s, n0 + 16 + lr, kc);
    v16bf a0 = frag_from_lds(a_s, m0 + lr, kc);
    v16bf a1 = frag_from_lds(a_s, m0 + 16 + lr, kc);
    acc00 = __builtin_amdgcn_wmma_f32_16x16x32_bf16(false, a0, false, b0, (short)0, acc00, false, false);
    acc01 = __builtin_amdgcn_wmma_f32_16x16x32_bf16(false, a0, false, b1, (short)0, acc01, false, false);
    acc10 = __builtin_amdgcn_wmma_f32_16x16x32_bf16(false, a1, false, b0, (short)0, acc10, false, false);
    acc11 = __builtin_amdgcn_wmma_f32_16x16x32_bf16(false, a1, false, b1, (short)0, acc11, false, false);
  }

  // C layout: vgpr v -> M row v (lanes 0-15) / v+8 (lanes 16-31); N = lane&15
  const int gn0 = blkN + n0 + lr;
  const int gn1 = blkN + n0 + 16 + lr;
#pragma unroll
  for (int v = 0; v < 8; ++v) {
    int row = v + ((lane >> 4) * 8);
    int gm0 = blkM + m0 + row;
    int gm1 = blkM + m0 + 16 + row;
    if (gm0 < M) {
      if (gn0 < N) Cs.put(gm0, gn0, acc00[v]);
      if (gn1 < N) Cs.put(gm0, gn1, acc01[v]);
    }
    if (gm1 < M) {
      if (gn0 < N) Cs.put(gm1, gn0, acc10[v]);
      if (gn1 < N) Cs.put(gm1, gn1, acc11[v]);
    }
  }
}

// ---------------- elementwise / glue kernels --------------------------
__global__ void k_f32_to_bf16(const float* in, __bf16* out, size_t n) {
  size_t i = (size_t)blockIdx.x * blockDim.x + threadIdx.x;
  if (i < n) out[i] = (__bf16)in[i];
}
__global__ void k_w_pad(const float* in, __bf16* out, int N, int K, int ld) {
  size_t i = (size_t)blockIdx.x * blockDim.x + threadIdx.x;
  if (i >= (size_t)N * ld) return;
  int n = (int)(i / ld), k = (int)(i - (size_t)n * ld);
  out[i] = (k < K) ? (__bf16)in[(size_t)n * K + k] : (__bf16)0.0f;
}
__global__ void k_maxpool(const __bf16* in, __bf16* out, int BC, int H, int W, int Ho, int Wo) {
  size_t i = (size_t)blockIdx.x * blockDim.x + threadIdx.x;
  size_t total = (size_t)BC * Ho * Wo;
  if (i >= total) return;
  int bc = (int)(i / (Ho * Wo)); int r = (int)(i - (size_t)bc * Ho * Wo);
  int oh = r / Wo, ow = r - oh * Wo;
  const __bf16* base = in + (size_t)bc * H * W + (oh * 2) * W + (ow * 2);
  float m = -1e30f;
#pragma unroll
  for (int dy = 0; dy < 3; ++dy)
#pragma unroll
    for (int dx = 0; dx < 3; ++dx)
      m = fmaxf(m, (float)base[dy * W + dx]);
  out[i] = (__bf16)m;
}
__global__ void k_row_norm(const __bf16* f, float* nrm, int M, int K) {
  int i = blockIdx.x * blockDim.x + threadIdx.x;
  if (i >= M) return;
  float s = 0.0f;
  const __bf16* row = f + (size_t)i * K;
  for (int k = 0; k < K; ++k) { float v = (float)row[k]; s += v * v; }
  nrm[i] = sqrtf(s);
}
__global__ void k_sim_scale(float* sim, const float* nrm, int B) {
  int i = blockIdx.x * blockDim.x + threadIdx.x;
  if (i >= B * B) return;
  int r = i / B, c = i - r * B;
  sim[i] = sim[i] / fmaxf(nrm[r] * nrm[c], 1e-8f);
}
__global__ void k_zero_f32(float* p, size_t n) {
  size_t i = (size_t)blockIdx.x * blockDim.x + threadIdx.x;
  if (i < n) p[i] = 0.0f;
}
__global__ void k_topk10(const float* sim, float* S, int B) {
  int i = blockIdx.x * blockDim.x + threadIdx.x;
  if (i >= B) return;
  unsigned mask[8] = {0, 0, 0, 0, 0, 0, 0, 0};
  const float* row = sim + (size_t)i * B;
  for (int t = 0; t < 10; ++t) {
    float best = -1e30f; int bidx = -1;
    for (int j = 0; j < B; ++j) {
      if (mask[j >> 5] & (1u << (j & 31))) continue;
      float v = row[j];
      if (v > best) { best = v; bidx = j; }
    }
    if (bidx >= 0) {
      mask[bidx >> 5] |= (1u << (bidx & 31));
      S[(size_t)i * B + bidx] = best;
    }
  }
}
__global__ void k_deg(const float* S, float* dinv, int B) {
  int i = blockIdx.x * blockDim.x + threadIdx.x;
  if (i >= B) return;
  float d = 0.0f;
  for (int j = 0; j < B; ++j) d += S[(size_t)i * B + j];
  dinv[i] = (d > 0.0f) ? rsqrtf(d) : 0.0f;
}
__global__ void k_degnorm(const float* S, const float* dinv, __bf16* Abf, int B) {
  int i = blockIdx.x * blockDim.x + threadIdx.x;
  if (i >= B * B) return;
  int r = i / B, c = i - r * B;
  Abf[i] = (__bf16)(dinv[r] * S[i] * dinv[c]);
}
__global__ void k_bn_stats(const float* hh, float* mu, float* var, int B, int C) {
  int c = blockIdx.x * blockDim.x + threadIdx.x;
  if (c >= C) return;
  float s = 0.0f, s2 = 0.0f;
  for (int b = 0; b < B; ++b) { float v = hh[(size_t)b * C + c]; s += v; s2 += v * v; }
  float m = s / B;
  mu[c] = m;
  var[c] = s2 / B - m * m;
}
__global__ void k_bn_apply(const float* hh, const float* mu, const float* var,
                           const float* g, const float* b, __bf16* out, int B, int C) {
  int i = blockIdx.x * blockDim.x + threadIdx.x;
  if (i >= B * C) return;
  int c = i % C;
  float v = (hh[i] - mu[c]) * rsqrtf(var[c] + 1e-5f) * g[c] + b[c];
  out[i] = (__bf16)v;
}

// =====================================================================
extern "C" void kernel_launch(void* const* d_in, const int* in_sizes, int n_in,
                              void* d_out, int out_size, void* d_ws, size_t ws_size,
                              hipStream_t stream) {
  (void)in_sizes; (void)n_in; (void)out_size; (void)ws_size;
  const int B = 256;

  // ---- inputs (setup_inputs order) ----
  const float* x    = (const float*)d_in[0];
  const float* cw1  = (const float*)d_in[1];  const float* cb1 = (const float*)d_in[2];
  const float* cw2  = (const float*)d_in[3];  const float* cb2 = (const float*)d_in[4];
  const float* cw3  = (const float*)d_in[5];  const float* cb3 = (const float*)d_in[6];
  const float* cw4  = (const float*)d_in[7];  const float* cb4 = (const float*)d_in[8];
  const float* cw5  = (const float*)d_in[9];  const float* cb5 = (const float*)d_in[10];
  const float* fcw  = (const float*)d_in[11]; const float* fcb = (const float*)d_in[12];
  const float* gw1  = (const float*)d_in[13]; const float* gb1 = (const float*)d_in[14];
  const float* gw2  = (const float*)d_in[15]; const float* gb2 = (const float*)d_in[16];
  const float* hw1  = (const float*)d_in[17]; const float* hb1 = (const float*)d_in[18];
  const float* bng  = (const float*)d_in[19]; const float* bnb = (const float*)d_in[20];
  const float* hw2  = (const float*)d_in[21]; const float* hb2 = (const float*)d_in[22];
  const float* cw   = (const float*)d_in[23]; const float* cb  = (const float*)d_in[24];
  float* out = (float*)d_out;                  // [256*64 code | 256*100 cla]

  // ---- workspace bump allocator (~295 MB total) ----
  size_t off = 0;
  auto alloc = [&](size_t bytes) -> void* {
    void* p = (char*)d_ws + off;
    off += (bytes + 255) & ~(size_t)255;
    return p;
  };
  const size_t NXB = (size_t)B * 3 * 224 * 224;
  const int LD1 = 368;                         // conv1 K=363 padded to 16n
  __bf16* xb   = (__bf16*)alloc(NXB * 2);
  __bf16* wc1b = (__bf16*)alloc((size_t)64 * LD1 * 2);
  __bf16* wc2b = (__bf16*)alloc((size_t)192 * 1600 * 2);
  __bf16* wc3b = (__bf16*)alloc((size_t)384 * 1728 * 2);
  __bf16* wc4b = (__bf16*)alloc((size_t)256 * 3456 * 2);
  __bf16* wc5b = (__bf16*)alloc((size_t)256 * 2304 * 2);
  __bf16* fcwb = (__bf16*)alloc((size_t)4096 * 9216 * 2);
  __bf16* gw1b = (__bf16*)alloc((size_t)1024 * 4096 * 2);
  __bf16* gw2b = (__bf16*)alloc((size_t)256 * 1024 * 2);
  __bf16* hw1b = (__bf16*)alloc((size_t)128 * 256 * 2);
  __bf16* hw2b = (__bf16*)alloc((size_t)64 * 128 * 2);
  __bf16* cwb  = (__bf16*)alloc((size_t)100 * 64 * 2);
  __bf16* P0   = (__bf16*)alloc((size_t)B * 64 * 55 * 55 * 2);   // 99 MB ping
  __bf16* P1   = (__bf16*)alloc((size_t)B * 64 * 27 * 27 * 2);   // 24 MB pong
  __bf16* F    = (__bf16*)alloc((size_t)B * 4096 * 2);
  __bf16* T1   = (__bf16*)alloc((size_t)B * 1024 * 2);
  __bf16* G1   = (__bf16*)alloc((size_t)B * 1024 * 2);
  __bf16* T2   = (__bf16*)alloc((size_t)B * 256 * 2);
  __bf16* G2   = (__bf16*)alloc((size_t)B * 256 * 2);
  __bf16* Abf  = (__bf16*)alloc((size_t)B * B * 2);
  __bf16* CODEB= (__bf16*)alloc((size_t)B * 64 * 2);
  __bf16* HHN  = (__bf16*)alloc((size_t)B * 128 * 2);
  float*  SIM  = (float*)alloc((size_t)B * B * 4);
  float*  S    = (float*)alloc((size_t)B * B * 4);
  float*  HH   = (float*)alloc((size_t)B * 128 * 4);
  float*  NRM  = (float*)alloc((size_t)B * 4);
  float*  DINV = (float*)alloc((size_t)B * 4);
  float*  MU   = (float*)alloc((size_t)128 * 4);
  float*  VAR  = (float*)alloc((size_t)128 * 4);

  auto cvt = [&](const float* src, __bf16* dst, size_t n) {
    k_f32_to_bf16<<<(unsigned)((n + 255) / 256), 256, 0, stream>>>(src, dst, n);
  };
  auto wpad = [&](const float* src, __bf16* dst, int N, int K, int ld) {
    size_t n = (size_t)N * ld;
    k_w_pad<<<(unsigned)((n + 255) / 256), 256, 0, stream>>>(src, dst, N, K, ld);
  };
  auto ggrid = [](int M, int N) { return dim3((unsigned)((N + 63) / 64), (unsigned)((M + 127) / 128)); };

  // ---- 1) bf16 conversions (weights row-padded to 16-elem stride) ----
  cvt(x, xb, NXB);
  wpad(cw1, wc1b, 64, 363, LD1);
  wpad(cw2, wc2b, 192, 1600, 1600);
  wpad(cw3, wc3b, 384, 1728, 1728);
  wpad(cw4, wc4b, 256, 3456, 3456);
  wpad(cw5, wc5b, 256, 2304, 2304);
  wpad(fcw, fcwb, 4096, 9216, 9216);
  wpad(gw1, gw1b, 1024, 4096, 4096);
  wpad(gw2, gw2b, 256, 1024, 1024);
  wpad(hw1, hw1b, 128, 256, 256);
  wpad(hw2, hw2b, 64, 128, 128);
  wpad(cw,  cwb,  100, 64, 64);

  // ---- 2) AlexNet features (implicit-GEMM WMMA convs + maxpools) ----
  { int M = B * 55 * 55, N = 64, K = 363;           // conv1: 224 -> 55
    gemm_wmma<<<ggrid(M, N), 256, 0, stream>>>(
      AConvT<3, 224, 224, 55, 55, 11, 4, 2>{xb}, BNK{wc1b, K, LD1},
      SConvT<64, 55 * 55>{P0, cb1}, M, N, K); }
  k_maxpool<<<(unsigned)(((size_t)B * 64 * 27 * 27 + 255) / 256), 256, 0, stream>>>(P0, P1, B * 64, 55, 55, 27, 27);

  { int M = B * 27 * 27, N = 192, K = 1600;         // conv2
    gemm_wmma<<<ggrid(M, N), 256, 0, stream>>>(
      AConvT<64, 27, 27, 27, 27, 5, 1, 2>{P1}, BNK{wc2b, K, K},
      SConvT<192, 27 * 27>{P0, cb2}, M, N, K); }
  k_maxpool<<<(unsigned)(((size_t)B * 192 * 13 * 13 + 255) / 256), 256, 0, stream>>>(P0, P1, B * 192, 27, 27, 13, 13);

  { int M = B * 13 * 13, N = 384, K = 1728;         // conv3
    gemm_wmma<<<ggrid(M, N), 256, 0, stream>>>(
      AConvT<192, 13, 13, 13, 13, 3, 1, 1>{P1}, BNK{wc3b, K, K},
      SConvT<384, 13 * 13>{P0, cb3}, M, N, K); }
  { int M = B * 13 * 13, N = 256, K = 3456;         // conv4
    gemm_wmma<<<ggrid(M, N), 256, 0, stream>>>(
      AConvT<384, 13, 13, 13, 13, 3, 1, 1>{P0}, BNK{wc4b, K, K},
      SConvT<256, 13 * 13>{P1, cb4}, M, N, K); }
  { int M = B * 13 * 13, N = 256, K = 2304;         // conv5
    gemm_wmma<<<ggrid(M, N), 256, 0, stream>>>(
      AConvT<256, 13, 13, 13, 13, 3, 1, 1>{P1}, BNK{wc5b, K, K},
      SConvT<256, 13 * 13>{P0, cb5}, M, N, K); }
  k_maxpool<<<(unsigned)(((size_t)B * 256 * 6 * 6 + 255) / 256), 256, 0, stream>>>(P0, P1, B * 256, 13, 13, 6, 6);

  // ---- 3) FC 9216 -> 4096 + relu ----
  gemm_wmma<<<ggrid(B, 4096), 256, 0, stream>>>(
    ADense{P1, 9216}, BNK{fcwb, 9216, 9216}, SDense<__bf16, 1>{F, fcb, 4096}, B, 4096, 9216);

  // ---- 4) cosine-similarity graph ----
  k_row_norm<<<1, 256, 0, stream>>>(F, NRM, B, 4096);
  gemm_wmma<<<ggrid(B, B), 256, 0, stream>>>(                      // f @ f^T
    ADense{F, 4096}, BNK{F, 4096, 4096}, SDense<float, 0>{SIM, nullptr, B}, B, B, 4096);
  k_sim_scale<<<(B * B + 255) / 256, 256, 0, stream>>>(SIM, NRM, B);
  k_zero_f32<<<(B * B + 255) / 256, 256, 0, stream>>>(S, (size_t)B * B);
  k_topk10<<<1, 256, 0, stream>>>(SIM, S, B);
  k_deg<<<1, 256, 0, stream>>>(S, DINV, B);
  k_degnorm<<<(B * B + 255) / 256, 256, 0, stream>>>(S, DINV, Abf, B);

  // ---- 5) 2-layer GCN ----
  gemm_wmma<<<ggrid(B, 1024), 256, 0, stream>>>(                   // f @ g_w1^T + b
    ADense{F, 4096}, BNK{gw1b, 4096, 4096}, SDense<__bf16, 0>{T1, gb1, 1024}, B, 1024, 4096);
  gemm_wmma<<<ggrid(B, 1024), 256, 0, stream>>>(                   // relu(A @ t1)
    ADense{Abf, B}, BKN{T1, 1024, B}, SDense<__bf16, 1>{G1, nullptr, 1024}, B, 1024, B);
  gemm_wmma<<<ggrid(B, 256), 256, 0, stream>>>(                    // g @ g_w2^T + b
    ADense{G1, 1024}, BNK{gw2b, 1024, 1024}, SDense<__bf16, 0>{T2, gb2, 256}, B, 256, 1024);
  gemm_wmma<<<ggrid(B, 256), 256, 0, stream>>>(                    // relu(A @ t2)
    ADense{Abf, B}, BKN{T2, 256, B}, SDense<__bf16, 1>{G2, nullptr, 256}, B, 256, B);

  // ---- 6) hash head: Linear+relu -> BN (batch stats) -> Linear ----
  gemm_wmma<<<ggrid(B, 128), 256, 0, stream>>>(
    ADense{G2, 256}, BNK{hw1b, 256, 256}, SDense<float, 1>{HH, hb1, 128}, B, 128, 256);
  k_bn_stats<<<1, 128, 0, stream>>>(HH, MU, VAR, B, 128);
  k_bn_apply<<<(B * 128 + 255) / 256, 256, 0, stream>>>(HH, MU, VAR, bng, bnb, HHN, B, 128);
  gemm_wmma<<<ggrid(B, 64), 256, 0, stream>>>(                     // code -> d_out[0:16384]
    ADense{HHN, 128}, BNK{hw2b, 128, 128}, SDense<float, 0>{out, hb2, 64}, B, 64, 128);

  // ---- 7) classifier: sigmoid(code @ c_w^T + b) -> d_out[16384:] ----
  cvt(out, CODEB, (size_t)B * 64);
  gemm_wmma<<<ggrid(B, 100), 256, 0, stream>>>(
    ADense{CODEB, 64}, BNK{cwb, 64, 64}, SDense<float, 2>{out + (size_t)B * 64, cb, 100}, B, 100, 64);
}
